// CWS_17712445129465
// MI455X (gfx1250) — matrix operand order repaired and checked by reading, not compiled, not run on hardware
//
#include <hip/hip_runtime.h>
#include <hip/hip_fp16.h>

typedef _Float16 v8h  __attribute__((ext_vector_type(8)));
typedef _Float16 v16h __attribute__((ext_vector_type(16)));
typedef float    v8f  __attribute__((ext_vector_type(8)));

#define Bsz   128
#define Slen  512
#define K0P   288      // 276 padded to multiple of 32
#define G4    1024     // 4*H gates
#define Hdim  256
#define NWG   16       // workgroups per direction in the scan kernel

__device__ __forceinline__ v16h cat8(v8h lo, v8h hi) {
  return __builtin_shufflevector(lo, hi, 0,1,2,3,4,5,6,7,8,9,10,11,12,13,14,15);
}
__device__ __forceinline__ float sigf(float x) { return 1.0f / (1.0f + __expf(-x)); }

// ---- CDNA5 async global<->LDS DMA (ASYNCcnt-tracked), inline asm ----------
__device__ __forceinline__ void async_g2l_b128(unsigned lds_byte, const void* gaddr) {
  asm volatile("global_load_async_to_lds_b128 %0, %1, off"
               :: "v"(lds_byte), "v"(gaddr) : "memory");
}
__device__ __forceinline__ void async_l2g_b128(void* gaddr, unsigned lds_byte) {
  asm volatile("global_store_async_from_lds_b128 %0, %1, off scope:SCOPE_DEV"
               :: "v"(gaddr), "v"(lds_byte) : "memory");
}
__device__ __forceinline__ void wait_async0() {
  asm volatile("s_wait_asynccnt 0x0" ::: "memory");
}

// ---------------------------------------------------------------------------
// Embedding gather: x0[b*S+s][0:288] = [word_emb | type_emb | zero pad], f16
// ---------------------------------------------------------------------------
__global__ void embed_kernel(const int* __restrict__ sentence,
                             const int* __restrict__ ctypes,
                             const float* __restrict__ wemb,
                             const float* __restrict__ temb,
                             _Float16* __restrict__ x0) {
  int r = blockIdx.x;            // b*512 + s
  int col = threadIdx.x;         // 0..287
  int wid = sentence[r];
  int ct  = ctypes[r];
  float v = 0.0f;
  if (col < 256)        v = wemb[(size_t)wid * 256 + col];
  else if (col < 276)   v = temb[ct * 20 + (col - 256)];
  x0[(size_t)r * K0P + col] = (_Float16)v;
}

// ---------------------------------------------------------------------------
// f32 -> f16 weight convert with column zero-padding (dst is rows x dc)
// ---------------------------------------------------------------------------
__global__ void cvt_pad(const float* __restrict__ src, _Float16* __restrict__ dst,
                        int rows, int sc, int dc) {
  int i = blockIdx.x * blockDim.x + threadIdx.x;
  if (i >= rows * dc) return;
  int row = i / dc, c = i - row * dc;
  dst[i] = (c < sc) ? (_Float16)src[(size_t)row * sc + c] : (_Float16)0.0f;
}

// ---------------------------------------------------------------------------
// Zero h double-buffers and scan barriers
// ---------------------------------------------------------------------------
__global__ void init_state(_Float16* __restrict__ hbuf, unsigned* __restrict__ bar) {
  int i = blockIdx.x * blockDim.x + threadIdx.x;
  if (i < 2 * 2 * Bsz * Hdim) hbuf[i] = (_Float16)0.0f;
  if (i < 2) bar[i] = 0u;
}

// ---------------------------------------------------------------------------
// WMMA GEMM: out = act(A(M x K) * W(N x K)^T + bias)
// M = gridDim.x*128, block tile 128x64, 8 waves, 2x2 wmma tiles/wave.
// Double-buffered LDS fed by async global->LDS DMA; epilogue staged through
// LDS and drained with async LDS->global stores (coalesced 16B).
// LDS arena (single object => group-segment offset 0):
//   A bufs:  [0, 20480)   2 x 128x40 halfs
//   W bufs:  [20480,30720) 2 x 64x40 halfs
//   C tile:  [30720,47104) 128x64 halfs
// ---------------------------------------------------------------------------
#define GA_OFF(b)  ((unsigned)((b) * 10240))
#define GW_OFF(b)  ((unsigned)(20480 + (b) * 5120))
#define GC_OFF     30720u

__global__ __launch_bounds__(256) void gemm_bias_f16(
    const _Float16* __restrict__ A, const _Float16* __restrict__ W,
    const float* __restrict__ bias, _Float16* __restrict__ out,
    int N, int K, int remap, int relu) {
  __shared__ __align__(16) char smem[47104];
  const int tid  = threadIdx.x;
  const int wave = tid >> 5, lane = tid & 31;
  const int half = lane >> 4, ln = lane & 15;
  const int m2 = wave & 3, n2 = wave >> 2;
  const int blockM = blockIdx.x * 128;
  const int blockN = blockIdx.y * 64;

  const int r0 = tid >> 2, q0 = tid & 3;     // A rows 0..63 / W rows 0..63
  const int r1 = r0 + 64;                    // A rows 64..127

  auto stage = [&](int kc, int bsel) {       // 3 async 16B copies per thread
    int k0 = kc << 5;
    async_g2l_b128(GA_OFF(bsel) + (unsigned)((r0 * 40 + q0 * 8) * 2),
                   &A[(size_t)(blockM + r0) * K + k0 + q0 * 8]);
    async_g2l_b128(GA_OFF(bsel) + (unsigned)((r1 * 40 + q0 * 8) * 2),
                   &A[(size_t)(blockM + r1) * K + k0 + q0 * 8]);
    async_g2l_b128(GW_OFF(bsel) + (unsigned)((r0 * 40 + q0 * 8) * 2),
                   &W[(size_t)(blockN + r0) * K + k0 + q0 * 8]);
  };

  v8f acc[2][2] = {};
  const int nk = K >> 5;
  stage(0, 0);
  for (int kc = 0; kc < nk; ++kc) {
    wait_async0();                 // chunk kc landed in LDS (my 3 copies)
    __syncthreads();               // everyone's copies landed
    const int cur = kc & 1;
    if (kc + 1 < nk) stage(kc + 1, cur ^ 1);   // overlaps with WMMA below

    const _Float16* Asc = (const _Float16*)(smem + GA_OFF(cur));
    const _Float16* Wsc = (const _Float16*)(smem + GW_OFF(cur));
    v16h a[2], b[2];
#pragma unroll
    for (int mi = 0; mi < 2; ++mi) {   // A frag: K = {8h..8h+7} U {8h+16..8h+23}
      int rowA = ((m2 * 2 + mi) << 4) + ln;
      v8h lo = *(const v8h*)&Asc[rowA * 40 + half * 8];
      v8h hi = *(const v8h*)&Asc[rowA * 40 + half * 8 + 16];
      a[mi] = cat8(lo, hi);
    }
#pragma unroll
    for (int ni = 0; ni < 2; ++ni) {   // B frag: K = {8h..8h+7} U {8h+16..8h+23}
      int ncol = ((n2 * 2 + ni) << 4) + ln;
      v8h lo = *(const v8h*)&Wsc[ncol * 40 + half * 8];
      v8h hi = *(const v8h*)&Wsc[ncol * 40 + half * 8 + 16];
      b[ni] = cat8(lo, hi);
    }
#pragma unroll
    for (int mi = 0; mi < 2; ++mi)
#pragma unroll
      for (int ni = 0; ni < 2; ++ni)
        acc[mi][ni] = __builtin_amdgcn_wmma_f32_16x16x32_f16(
            false, a[mi], false, b[ni], (short)0, acc[mi][ni], false, false);
  }

  // epilogue: stage C tile in LDS, drain with coalesced async stores
  _Float16* Cs = (_Float16*)(smem + GC_OFF);
#pragma unroll
  for (int mi = 0; mi < 2; ++mi)
#pragma unroll
    for (int ni = 0; ni < 2; ++ni) {
      int coll = ((n2 * 2 + ni) << 4) + ln;
      float bv = bias[blockN + coll];
#pragma unroll
      for (int r = 0; r < 8; ++r) {
        int rowl = ((m2 * 2 + mi) << 4) + r + half * 8;
        float v = acc[mi][ni][r] + bv;
        if (relu) v = fmaxf(v, 0.0f);
        Cs[rowl * 64 + coll] = (_Float16)v;
      }
    }
  __syncthreads();
#pragma unroll
  for (int i = 0; i < 4; ++i) {        // 1024 x 16B chunks, 4 per thread
    int idx = tid + i * 256;
    int row = idx >> 3, q = idx & 7;
    int rowg = blockM + row;
    size_t orow = remap ? (size_t)(((rowg & 511) << 7) + (rowg >> 9))
                        : (size_t)rowg;
    async_l2g_b128(&out[orow * (size_t)N + blockN + q * 8],
                   GC_OFF + (unsigned)((row * 64 + q * 8) * 2));
  }
  wait_async0();
}

// ---------------------------------------------------------------------------
// Cooperative bidirectional LSTM scan for one layer.
// grid = (NWG, 2); workgroup wg of direction d owns h columns [16*wg,16*wg+16).
// Whh slice resident in LDS; per-step xw gate tile async-staged into LDS
// overlapping the recurrence WMMAs; h slice drained via async LDS->global.
// LDS arena:
//   Whh:  [0, 33792)      64 x 264 halfs (packed p = gate*16+n)
//   XW:   [33792, 50176)  128 x 64 halfs (4 gates x 16 cols)
//   HST:  [50176, 54272)  128 x 16 halfs (h slice staging)
//   LEN:  [54272, 54784)  128 ints
// ---------------------------------------------------------------------------
#define SW_OFF  0u
#define SX_OFF  33792u
#define SH_OFF  50176u
#define SL_OFF  54272u

__global__ __launch_bounds__(256) void lstm_scan(
    const _Float16* __restrict__ xwF, const _Float16* __restrict__ xwB,
    const _Float16* __restrict__ whhF, const _Float16* __restrict__ whhB,
    _Float16* __restrict__ hbuf,       // [2][2][128][256]
    _Float16* __restrict__ hcat,       // [B*S][512]
    const int* __restrict__ length,
    unsigned int* __restrict__ bar) {  // [2]
  __shared__ __align__(16) char smem[54784];
  const int dir = blockIdx.y;
  const int wg  = blockIdx.x;
  const _Float16* xw  = dir ? xwB : xwF;
  const _Float16* whh = dir ? whhB : whhF;
  _Float16* hb = hbuf + (size_t)dir * 2 * Bsz * Hdim;
  unsigned* mybar = bar + dir;

  const _Float16* Ws  = (const _Float16*)(smem + SW_OFF);
  const _Float16* XWs = (const _Float16*)(smem + SX_OFF);
  _Float16*       hst = (_Float16*)(smem + SH_OFF);
  int*            lens_s = (int*)(smem + SL_OFF);

  const int tid = threadIdx.x;
  for (int it = 0; it < 8; ++it) {     // Whh slice: 64 rows x 32 x 16B, async
    int idx = it * 256 + tid;
    int p = idx >> 5, q = idx & 31;
    int grow = (p >> 4) * Hdim + wg * 16 + (p & 15);   // global gate row
    async_g2l_b128(SW_OFF + (unsigned)((p * 264 + q * 8) * 2),
                   &whh[(size_t)grow * Hdim + q * 8]);
  }
  if (tid < Bsz) lens_s[tid] = length[tid];
  wait_async0();
  __syncthreads();

  const int wave = tid >> 5, lane = tid & 31;
  const int half = lane >> 4, ln = lane & 15;
  const int rowA = wave * 16 + ln;     // A-frag batch row
  int lenr[8];
#pragma unroll
  for (int r = 0; r < 8; ++r) lenr[r] = lens_s[wave * 16 + r + half * 8];

  float c[8], hkeep[8];
#pragma unroll
  for (int r = 0; r < 8; ++r) { c[r] = 0.0f; hkeep[r] = 0.0f; }

  const int rs = tid >> 1, qs = tid & 1;   // h-slice drain mapping

  for (int t = 0; t < Slen; ++t) {
    const _Float16* rd = hb + (size_t)(t & 1) * Bsz * Hdim;
    _Float16*       wr = hb + (size_t)((t + 1) & 1) * Bsz * Hdim;

    // stage this step's xw gate tile (overlaps with WMMAs below)
#pragma unroll
    for (int i = 0; i < 4; ++i) {
      int idx = tid + i * 256;           // 1024 x 16B chunks
      int row = idx >> 3, gate = (idx >> 1) & 3, q = idx & 1;
      int len = lens_s[row];
      int pos = dir ? ((t < len) ? (len - 1 - t) : t) : t;
      async_g2l_b128(SX_OFF + (unsigned)((row * 64 + gate * 16 + q * 8) * 2),
                     xw + ((size_t)pos * Bsz + row) * G4 + gate * Hdim + wg * 16 + q * 8);
    }

    v8f acc[4] = {};
#pragma unroll
    for (int kk = 0; kk < 8; ++kk) {
      v8h alo = *(const v8h*)&rd[rowA * Hdim + kk * 32 + half * 8];
      v8h ahi = *(const v8h*)&rd[rowA * Hdim + kk * 32 + half * 8 + 16];
      v16h a = cat8(alo, ahi);
#pragma unroll
      for (int nt = 0; nt < 4; ++nt) {
        int p = nt * 16 + ln;
        v8h blo = *(const v8h*)&Ws[p * 264 + kk * 32 + half * 8];
        v8h bhi = *(const v8h*)&Ws[p * 264 + kk * 32 + half * 8 + 16];
        acc[nt] = __builtin_amdgcn_wmma_f32_16x16x32_f16(
            false, a, false, cat8(blo, bhi), (short)0, acc[nt], false, false);
      }
    }

    wait_async0();                  // xw tile landed (mine)
    __syncthreads();                // everyone's

#pragma unroll
    for (int r = 0; r < 8; ++r) {
      int row = wave * 16 + r + half * 8;          // batch index
      int len = lenr[r];
      float gi = acc[0][r] + (float)XWs[row * 64 +      ln];
      float gf = acc[1][r] + (float)XWs[row * 64 + 16 + ln];
      float gg = acc[2][r] + (float)XWs[row * 64 + 32 + ln];
      float go = acc[3][r] + (float)XWs[row * 64 + 48 + ln];
      gi = sigf(gi); gf = sigf(gf); go = sigf(go); gg = tanhf(gg);
      float cn = gf * c[r] + gi * gg;
      float hn = go * tanhf(cn);
      bool upd = (t < len);
      c[r]     = upd ? cn : c[r];
      hkeep[r] = upd ? hn : hkeep[r];
      hst[row * 16 + ln] = (_Float16)hkeep[r];
    }
    __syncthreads();

    // drain h slice: 256 x 16B chunks -> h double-buffer + hcat output
    {
      int len = lens_s[rs];
      int pos = dir ? ((t < len) ? (len - 1 - t) : t) : t;
      unsigned loff = SH_OFF + (unsigned)((rs * 16 + qs * 8) * 2);
      async_l2g_b128(&wr[rs * Hdim + wg * 16 + qs * 8], loff);
      async_l2g_b128(&hcat[((size_t)rs * Slen + pos) * 512 + dir * Hdim + wg * 16 + qs * 8],
                     loff);
    }
    wait_async0();
    __syncthreads();
    if (tid == 0) {                    // device-scope step barrier (per dir)
      __threadfence();
      atomicAdd(mybar, 1u);
      unsigned target = (unsigned)(t + 1) * gridDim.x;
      while (__hip_atomic_load(mybar, __ATOMIC_ACQUIRE,
                               __HIP_MEMORY_SCOPE_AGENT) < target)
        __builtin_amdgcn_s_sleep(1);
      __threadfence();
    }
    __syncthreads();
  }
}

// ---------------------------------------------------------------------------
// emissions[r][t] = b2[t] + sum_k hid[r][k] * W2[t][k]   (T=4, K=512)
// ---------------------------------------------------------------------------
__global__ void emis_kernel(const _Float16* __restrict__ hid,
                            const float* __restrict__ W2,
                            const float* __restrict__ b2,
                            float* __restrict__ emis) {
  int i = blockIdx.x * blockDim.x + threadIdx.x;   // < 65536*4
  int r = i >> 2, tt = i & 3;
  const v8h* h8 = (const v8h*)(hid + (size_t)r * 512);
  const float* w = W2 + tt * 512;
  float s = b2[tt];
  for (int k = 0; k < 64; ++k) {
    v8h hv = h8[k];
#pragma unroll
    for (int j = 0; j < 8; ++j) s += (float)hv[j] * w[k * 8 + j];
  }
  emis[i] = s;
}

// ---------------------------------------------------------------------------
// CRF NLL: one lane per batch, logsumexp forward scan, mean over batch.
// ---------------------------------------------------------------------------
__global__ __launch_bounds__(128) void crf_kernel(
    const float* __restrict__ emis, const int* __restrict__ tags,
    const int* __restrict__ length, const float* __restrict__ start,
    const float* __restrict__ endt, const float* __restrict__ trans,
    float* __restrict__ out) {
  __shared__ float red[128];
  int b = threadIdx.x;
  int len = length[b];
  const int* tg = tags + b * Slen;
  const float* eb = emis + (size_t)b * Slen * 4;

  float num = start[tg[0]];
  for (int s = 0; s < len; ++s) num += eb[s * 4 + tg[s]];
  for (int s = 1; s < len; ++s) num += trans[tg[s - 1] * 4 + tg[s]];
  num += endt[tg[len - 1]];

  float al[4];
#pragma unroll
  for (int j = 0; j < 4; ++j) al[j] = start[j] + eb[j];
  for (int s = 1; s < len; ++s) {
    float nx[4];
#pragma unroll
    for (int j = 0; j < 4; ++j) {
      float m = -1e30f;
#pragma unroll
      for (int i = 0; i < 4; ++i) m = fmaxf(m, al[i] + trans[i * 4 + j]);
      float sum = 0.0f;
#pragma unroll
      for (int i = 0; i < 4; ++i) sum += __expf(al[i] + trans[i * 4 + j] - m);
      nx[j] = m + __logf(sum) + eb[s * 4 + j];
    }
#pragma unroll
    for (int j = 0; j < 4; ++j) al[j] = nx[j];
  }
  float m = -1e30f;
#pragma unroll
  for (int j = 0; j < 4; ++j) m = fmaxf(m, al[j] + endt[j]);
  float sum = 0.0f;
#pragma unroll
  for (int j = 0; j < 4; ++j) sum += __expf(al[j] + endt[j] - m);
  float denom = m + __logf(sum);

  red[b] = denom - num;                 // -llh
  __syncthreads();
  for (int s2 = 64; s2 > 0; s2 >>= 1) {
    if (b < s2) red[b] += red[b + s2];
    __syncthreads();
  }
  if (b == 0) out[0] = red[0] * (1.0f / 128.0f);
}

// ---------------------------------------------------------------------------
extern "C" void kernel_launch(void* const* d_in, const int* in_sizes, int n_in,
                              void* d_out, int out_size, void* d_ws, size_t ws_size,
                              hipStream_t stream) {
  (void)in_sizes; (void)n_in; (void)out_size; (void)ws_size;
  // setup_inputs() dict order; lstm_params in insertion order l0f,l0b,l1f,l1b
  const int*   sentence = (const int*)d_in[0];
  const int*   tags     = (const int*)d_in[1];
  const int*   ctypes   = (const int*)d_in[2];
  /* d_in[3] = mask: redundant with length, unused */
  const int*   length   = (const int*)d_in[4];
  const float* wemb     = (const float*)d_in[5];
  const float* temb     = (const float*)d_in[6];
  const float* Wih0f = (const float*)d_in[7];
  const float* Whh0f = (const float*)d_in[8];
  const float* bi0f  = (const float*)d_in[9];
  const float* Wih0b = (const float*)d_in[10];
  const float* Whh0b = (const float*)d_in[11];
  const float* bi0b  = (const float*)d_in[12];
  const float* Wih1f = (const float*)d_in[13];
  const float* Whh1f = (const float*)d_in[14];
  const float* bi1f  = (const float*)d_in[15];
  const float* Wih1b = (const float*)d_in[16];
  const float* Whh1b = (const float*)d_in[17];
  const float* bi1b  = (const float*)d_in[18];
  const float* W1 = (const float*)d_in[19];
  const float* b1 = (const float*)d_in[20];
  const float* W2 = (const float*)d_in[21];
  const float* b2 = (const float*)d_in[22];
  const float* st = (const float*)d_in[23];
  const float* et = (const float*)d_in[24];
  const float* tr = (const float*)d_in[25];

  // workspace carve (peak ~450 MB)
  size_t off = 0;
  char* base = (char*)d_ws;
  auto carve = [&](size_t bytes) -> void* {
    void* p = base + off;
    off += (bytes + 255) & ~(size_t)255;
    return p;
  };
  _Float16* x0   = (_Float16*)carve((size_t)65536 * K0P * 2);
  _Float16* w0f  = (_Float16*)carve((size_t)G4 * K0P * 2);
  _Float16* w0b  = (_Float16*)carve((size_t)G4 * K0P * 2);
  _Float16* u0f  = (_Float16*)carve((size_t)G4 * Hdim * 2);
  _Float16* u0b  = (_Float16*)carve((size_t)G4 * Hdim * 2);
  _Float16* w1f  = (_Float16*)carve((size_t)G4 * 512 * 2);
  _Float16* w1b  = (_Float16*)carve((size_t)G4 * 512 * 2);
  _Float16* u1f  = (_Float16*)carve((size_t)G4 * Hdim * 2);
  _Float16* u1b  = (_Float16*)carve((size_t)G4 * Hdim * 2);
  _Float16* wW1  = (_Float16*)carve((size_t)512 * 512 * 2);
  _Float16* xwA  = (_Float16*)carve((size_t)Slen * Bsz * G4 * 2);   // 134 MB
  _Float16* xwB  = (_Float16*)carve((size_t)Slen * Bsz * G4 * 2);   // 134 MB
  _Float16* h0   = (_Float16*)carve((size_t)65536 * 512 * 2);
  _Float16* h1   = (_Float16*)carve((size_t)65536 * 512 * 2);
  _Float16* hbuf = (_Float16*)carve((size_t)2 * 2 * Bsz * Hdim * 2);
  unsigned* bar  = (unsigned*)carve(256);
  _Float16* hid2 = xwA;                  // alias: xw dead after layer-1 scan
  float*    emis = (float*)xwB;          // alias

  auto cv = [&](const float* s, _Float16* d, int rows, int sc, int dc) {
    int n = rows * dc;
    hipLaunchKernelGGL(cvt_pad, dim3((n + 255) / 256), dim3(256), 0, stream,
                       s, d, rows, sc, dc);
  };
  cv(Wih0f, w0f, G4, 276, K0P);  cv(Wih0b, w0b, G4, 276, K0P);
  cv(Whh0f, u0f, G4, Hdim, Hdim); cv(Whh0b, u0b, G4, Hdim, Hdim);
  cv(Wih1f, w1f, G4, 512, 512);  cv(Wih1b, w1b, G4, 512, 512);
  cv(Whh1f, u1f, G4, Hdim, Hdim); cv(Whh1b, u1b, G4, Hdim, Hdim);
  cv(W1, wW1, 512, 512, 512);

  hipLaunchKernelGGL(embed_kernel, dim3(65536), dim3(K0P), 0, stream,
                     sentence, ctypes, wemb, temb, x0);

  dim3 gX(512, 16);   // 65536x1024 GEMMs
  hipLaunchKernelGGL(gemm_bias_f16, gX, dim3(256), 0, stream,
                     x0, w0f, bi0f, xwA, G4, K0P, 1, 0);
  hipLaunchKernelGGL(gemm_bias_f16, gX, dim3(256), 0, stream,
                     x0, w0b, bi0b, xwB, G4, K0P, 1, 0);
  hipLaunchKernelGGL(init_state, dim3(512), dim3(256), 0, stream, hbuf, bar);
  hipLaunchKernelGGL(lstm_scan, dim3(NWG, 2), dim3(256), 0, stream,
                     xwA, xwB, u0f, u0b, hbuf, h0, length, bar);

  hipLaunchKernelGGL(gemm_bias_f16, gX, dim3(256), 0, stream,
                     h0, w1f, bi1f, xwA, G4, 512, 1, 0);
  hipLaunchKernelGGL(gemm_bias_f16, gX, dim3(256), 0, stream,
                     h0, w1b, bi1b, xwB, G4, 512, 1, 0);
  hipLaunchKernelGGL(init_state, dim3(512), dim3(256), 0, stream, hbuf, bar);
  hipLaunchKernelGGL(lstm_scan, dim3(NWG, 2), dim3(256), 0, stream,
                     xwA, xwB, u1f, u1b, hbuf, h1, length, bar);

  dim3 gF(512, 8);    // FFN: 65536x512, ReLU
  hipLaunchKernelGGL(gemm_bias_f16, gF, dim3(256), 0, stream,
                     h1, wW1, b1, hid2, 512, 512, 0, 1);
  hipLaunchKernelGGL(emis_kernel, dim3(1024), dim3(256), 0, stream,
                     hid2, W2, b2, emis);
  hipLaunchKernelGGL(crf_kernel, dim3(1), dim3(128), 0, stream,
                     emis, tags, length, st, et, tr, (float*)d_out);
}
